// SCNN_68822555951190
// MI455X (gfx1250) — matrix operand order repaired and checked by reading, not compiled, not run on hardware
//
#include <hip/hip_runtime.h>
#include <hip/hip_bf16.h>

// ---------------------------------------------------------------------------
// SCNN message passing on MI455X (gfx1250), bf16 WMMA with f32 accumulation.
// Each directional scan step is the GEMM
//   out[co,n] = sum_{ci,k} y_prev[ci, n+k-4] * W[co,ci,k]
//   M = 128 (C_out), K = 1152 (C_in * 9 taps), N = 256 (W) or 128 (H).
// Carried state y_prev lives in dynamic LDS as bf16, column-major over n with
// 4 zero halo columns each side (SAME padding -> branchless inner loop),
// XOR-block-swizzled for bank spread.
// Contraction is tap-major so each K=32 WMMA chunk has a fixed tap ->
// B operand per lane is one contiguous 32B ci-run (2x ds_load_b128),
// A operand is 8 immediate-offset global_load_b128 in one clause per tap.
// ---------------------------------------------------------------------------

typedef __attribute__((ext_vector_type(16))) __bf16        v16bf;
typedef __attribute__((ext_vector_type(8)))  float         v8f;
typedef __attribute__((ext_vector_type(4)))  float         f32x4;
typedef __attribute__((ext_vector_type(4)))  unsigned int  u32x4;
typedef __attribute__((ext_vector_type(8)))  unsigned int  u32x8;

__device__ __forceinline__ unsigned short f2bf(float f) {
    return (unsigned short)((__float_as_uint(f) + 0x8000u) >> 16);
}

__device__ __forceinline__ v16bf make_v16bf(u32x4 lo, u32x4 hi) {
    u32x8 w = __builtin_shufflevector(lo, hi, 0, 1, 2, 3, 4, 5, 6, 7);
    return __builtin_bit_cast(v16bf, w);
}

// XOR block swizzle: 16-element (32B) ci-blocks permuted per column.
// phys(ci, cpad) = ((ci>>4) ^ (cpad&7))*16 + (ci&15); a 16-aligned ci-run of
// 16 stays contiguous and ascending; banks differ across consecutive columns.
__device__ __forceinline__ int physidx(int ci, int cpad) {
    return ((((ci >> 4) ^ (cpad & 7)) << 4) | (ci & 15));
}

// ---------------------------------------------------------------------------
// Weight repack: w[co][ci][k] f32  ->  wp[k][co][ci] bf16  (147456 elements)
// ---------------------------------------------------------------------------
__global__ void pack_weights(const float* __restrict__ w,
                             unsigned short* __restrict__ wp) {
    int i = blockIdx.x * 256 + threadIdx.x;
    if (i >= 9 * 128 * 128) return;
    int k  = i / (128 * 128);
    int r  = i - k * 128 * 128;
    int co = r >> 7;
    int ci = r & 127;
    wp[i] = f2bf(w[((co << 7) + ci) * 9 + k]);
}

// ---------------------------------------------------------------------------
// One directional scan pass. Grid = B workgroups, 1024 threads = 32 waves.
// Wave -> (mTile = wid&7, nGroup = wid>>3); each nGroup owns NTG 16-wide
// N tiles. N in {256,128}, NTG = N/64. SSTR/NSTR are compile-time strides.
// Dynamic LDS: (N+8) columns x 128 ci, bf16.
// ---------------------------------------------------------------------------
template <int N, int NTG, int SSTR, int NSTR>
__global__ __launch_bounds__(1024, 1)
void scan_pass(const float* __restrict__ xin, float* __restrict__ yout,
               const unsigned short* __restrict__ wp, int L, int reverse) {
    constexpr int       NSH  = (N == 256) ? 8 : 7;   // log2(N)
    constexpr long long CSTR = 32768;                // H*W
    constexpr long long BSTR = 4194304;              // C*H*W

    extern __shared__ unsigned short yT[];           // (N+8)*128 bf16

    const int tid  = threadIdx.x;
    const int lane = tid & 31;
    const int nL   = lane & 15;
    const int half = lane >> 4;
    // wave-uniform ids, forced into SGPRs
    const int wid    = __builtin_amdgcn_readfirstlane(tid >> 5);
    const int mTile  = wid & 7;
    const int nGroup = wid >> 3;
    const int nBase  = (nGroup * NTG) << 4;

    const float* xb = xin + (long long)blockIdx.x * BSTR;
    float*       yb = yout + (long long)blockIdx.x * BSTR;

    // Per-wave weight base: wp[k][co][ci], co fixed per lane for A operand.
    const unsigned short* wbase = wp + ((mTile << 4) + nL) * 128 + (half << 4);

    // ---- zero the 8 halo columns (never written again) -------------------
    for (int i = tid; i < 8 * 128; i += 1024) {
        int g    = i >> 7;                        // 0..7
        int cpad = (g < 4) ? g : (N + g);         // 0..3, N+4..N+7
        yT[(cpad << 7) + (i & 127)] = 0;
    }

    // ---- step 0: first slice passes through; seed LDS state --------------
    {
        const int sIdx = reverse ? (L - 1) : 0;
        const float* xs = xb + (long long)sIdx * SSTR;
        float*       ys = yb + (long long)sIdx * SSTR;
        if constexpr (NSTR == 1) {
            for (int i = tid; i < (N * 128) / 4; i += 1024) {
                int c  = i >> (NSH - 2);
                int n4 = (i & ((N >> 2) - 1)) << 2;
                long long a = (long long)c * CSTR + n4;
                f32x4 v = *(const f32x4*)(xs + a);
                *(f32x4*)(ys + a) = v;
#pragma unroll
                for (int j = 0; j < 4; ++j) {
                    int cpad = n4 + j + 4;
                    yT[(cpad << 7) + physidx(c, cpad)] = f2bf(v[j]);
                }
            }
        } else {
            for (int i = tid; i < N * 128; i += 1024) {
                int c    = i >> NSH;
                int n    = i & (N - 1);
                int cpad = n + 4;
                long long a = (long long)c * CSTR + (long long)n * NSTR;
                float v = xs[a];
                ys[a]   = v;
                yT[(cpad << 7) + physidx(c, cpad)] = f2bf(v);
            }
        }
    }
    __syncthreads();

    // ---- remaining steps --------------------------------------------------
    for (int s = 1; s < L; ++s) {
        const int sIdx = reverse ? (L - 1 - s) : s;
        const float* xs = xb + (long long)sIdx * SSTR;
        float*       ys = yb + (long long)sIdx * SSTR;
        const int co0 = (mTile << 4) + (half << 3);

        // Preload this step's x (hides under the GEMM); prefetch next step's.
        const long long sdir = (long long)(reverse ? -SSTR : SSTR);
        float xv[NTG][8];
#pragma unroll
        for (int t = 0; t < NTG; ++t) {
            long long a0 = (long long)co0 * CSTR +
                           (long long)(nBase + (t << 4) + nL) * NSTR;
#pragma unroll
            for (int r = 0; r < 8; ++r) xv[t][r] = xs[a0 + r * CSTR];
            if (s + 1 < L) __builtin_prefetch(xs + sdir + a0, 0, 1);
        }

        v8f acc[NTG];
#pragma unroll
        for (int t = 0; t < NTG; ++t)
            acc[t] = (v8f){0.f, 0.f, 0.f, 0.f, 0.f, 0.f, 0.f, 0.f};

        const int cbase = nBase + nL;                // cpad = cbase + t*16 + k
        const unsigned short* wk = wbase;
#pragma unroll 1
        for (int k = 0; k < 9; ++k) {
            // A operands for this tap: 4 ci-chunks, shared by all N tiles.
            v16bf A[4];
#pragma unroll
            for (int c = 0; c < 4; ++c) {
                int ci0 = c << 5;
                u32x4 a0 = *(const u32x4*)(wk + ci0);
                u32x4 a1 = *(const u32x4*)(wk + ci0 + 8);
                A[c] = make_v16bf(a0, a1);
            }
            wk += 16384;                             // next tap plane (+32KB)

#pragma unroll
            for (int t = 0; t < NTG; ++t) {
                const int cpad = cbase + (t << 4) + k;   // halo handles edges
                const unsigned short* brow = yT + (cpad << 7);
                const int x8 = cpad & 7;
#pragma unroll
                for (int c = 0; c < 4; ++c) {
                    const unsigned short* bp =
                        brow + ((((c << 1) | half) ^ x8) << 4);
                    u32x4 b0 = *(const u32x4*)bp;
                    u32x4 b1 = *(const u32x4*)(bp + 8);
                    acc[t] = __builtin_amdgcn_wmma_f32_16x16x32_bf16(
                        false, A[c], false, make_v16bf(b0, b1),
                        (short)0, acc[t], false, false);
                }
            }
        }

        // Epilogue: y = x + relu(conv); write global, pack bf16 pairs for LDS
        // via v_perm_b32 (one op packs two rounded high-halves).
        unsigned int pk[NTG][4];
#pragma unroll
        for (int t = 0; t < NTG; ++t) {
            long long a0 = (long long)co0 * CSTR +
                           (long long)(nBase + (t << 4) + nL) * NSTR;
#pragma unroll
            for (int p = 0; p < 4; ++p) {
                float c0 = acc[t][2 * p];
                float c1 = acc[t][2 * p + 1];
                c0 = c0 > 0.f ? c0 : 0.f;
                c1 = c1 > 0.f ? c1 : 0.f;
                float y0 = xv[t][2 * p] + c0;
                float y1 = xv[t][2 * p + 1] + c1;
                ys[a0 + (2 * p) * CSTR]     = y0;
                ys[a0 + (2 * p + 1) * CSTR] = y1;
                unsigned u0 = __float_as_uint(y0) + 0x8000u;
                unsigned u1 = __float_as_uint(y1) + 0x8000u;
                pk[t][p] = __builtin_amdgcn_perm(u1, u0, 0x07060302u);
            }
        }

        __syncthreads();   // all waves finished reading yT

        // State write-back: one ds_store_b128 per tile per lane.
        // co = mTile*16 + half*8 + r  ->  block = mTile, inblk = half*8+r.
#pragma unroll
        for (int t = 0; t < NTG; ++t) {
            int cpad = nBase + (t << 4) + nL + 4;
            int pbas = ((mTile ^ (cpad & 7)) << 4) + (half << 3);
            u32x4 st = {pk[t][0], pk[t][1], pk[t][2], pk[t][3]};
            *(u32x4*)(yT + (cpad << 7) + pbas) = st;
        }
        __syncthreads();   // new state visible
    }
}

// ---------------------------------------------------------------------------
// Launch: pack weights, then 4 chained directional passes.
//   pass1 (down):  x    -> ws0
//   pass2 (up):    ws0  -> ws0  (in-place safe: recurrent operand is in LDS)
//   pass3 (right): ws0  -> out
//   pass4 (left):  out  -> out
// ---------------------------------------------------------------------------
extern "C" void kernel_launch(void* const* d_in, const int* in_sizes, int n_in,
                              void* d_out, int out_size, void* d_ws, size_t ws_size,
                              hipStream_t stream) {
    (void)in_sizes; (void)n_in; (void)out_size; (void)ws_size;

    const int B = 16, H = 128, W = 256;

    const float* x    = (const float*)d_in[0];
    const float* w_ud = (const float*)d_in[1];
    const float* w_du = (const float*)d_in[2];
    const float* w_lr = (const float*)d_in[3];
    const float* w_rl = (const float*)d_in[4];

    const size_t WPACK = (size_t)9 * 128 * 128;      // elements per direction
    unsigned short* wp_ud = (unsigned short*)d_ws;
    unsigned short* wp_du = wp_ud + WPACK;
    unsigned short* wp_lr = wp_du + WPACK;
    unsigned short* wp_rl = wp_lr + WPACK;
    float* ws0 = (float*)((char*)d_ws + 4 * WPACK * sizeof(unsigned short));

    const int pgrid = (int)((WPACK + 255) / 256);
    pack_weights<<<pgrid, 256, 0, stream>>>(w_ud, wp_ud);
    pack_weights<<<pgrid, 256, 0, stream>>>(w_du, wp_du);
    pack_weights<<<pgrid, 256, 0, stream>>>(w_lr, wp_lr);
    pack_weights<<<pgrid, 256, 0, stream>>>(w_rl, wp_rl);

    float* out = (float*)d_out;

    const size_t smemW = (size_t)(256 + 8) * 128 * sizeof(unsigned short);
    const size_t smemH = (size_t)(128 + 8) * 128 * sizeof(unsigned short);

    // down/up: scan over H (L=128), slice width N = W = 256, n contiguous
    scan_pass<256, 4, 256, 1><<<B, 1024, smemW, stream>>>(x,   ws0, wp_ud, H, 0);
    scan_pass<256, 4, 256, 1><<<B, 1024, smemW, stream>>>(ws0, ws0, wp_du, H, 1);
    // right/left: scan over W (L=256), slice width N = H = 128, n stride W
    scan_pass<128, 2, 1, 256><<<B, 1024, smemH, stream>>>(ws0, out, wp_lr, W, 0);
    scan_pass<128, 2, 1, 256><<<B, 1024, smemH, stream>>>(out, out, wp_rl, W, 1);
}
